// MultiHeadAttention_75273596829862
// MI455X (gfx1250) — compile-verified, matrix operand order checked by reading, not run on hardware
//
#include <hip/hip_runtime.h>
#include <hip/hip_bf16.h>

// ---------------------------------------------------------------------------
// MHA for MI455X (gfx1250, wave32, WMMA). Strategy:
//  1. One-time fp32->bf16 conversion of activations, and transpose-convert of
//     weights to [n][k] bf16. All bf16 operands (X=6.3MB, Wt=1.2MB, K+V=12.6MB)
//     are L2-resident (192MB L2), so...
//  2. ...all GEMMs are LDS-free, barrier-free global(L2)->WMMA streams using
//     v_wmma_f32_16x16x32_bf16, 8 WMMA per wave per 32-wide k-step.
//  3. Fused flash attention: per-wave 16 queries, streams 32-key tiles,
//     online softmax via width-16 xor shuffles matching the C-layout rows.
// HBM traffic ~ first touch only (~60MB, ~2.6us @23.3TB/s); compute ~45 GFLOP
// on bf16 matrix cores -> pipeline sits at the WMMA roofline.
//
// Workspace (P = 4096*768*2 B = 6.29MB, WSZ = 768*768*2 B = 1.18MB):
//   [0P] q[B,H,S,hd]  [1P] k[B,H,S,hd]  [2P] vT[B,H,hd,S]
//   [3P] xq (later reused as attn[tok,D])   [4P] xk   [5P] xv
//   [6P] wqt [6P+1W] wkt [6P+2W] wvt [6P+3W] wot      (all [n][k] bf16)
// ---------------------------------------------------------------------------

typedef __bf16 bf16_t;
typedef bf16_t v16bf __attribute__((ext_vector_type(16)));
typedef float  v8f   __attribute__((ext_vector_type(8)));

constexpr int NB     = 2;
constexpr int SEQ    = 2048;
constexpr int NH     = 12;
constexpr int DMODEL = 768;
constexpr int HD     = 64;
constexpr int NTOK   = NB * SEQ;   // 4096

union FragBF {                 // 16x32 bf16 fragment = 8 VGPRs
    uint4 u[2];
    v16bf v;
};

__device__ __forceinline__ v8f wmma_bf16(v16bf a, v16bf b, v8f c) {
    return __builtin_amdgcn_wmma_f32_16x16x32_bf16(
        false, a, false, b, (short)0, c, false, false);
}

__device__ __forceinline__ uint4 ld16g(const bf16_t* p) {
    return *reinterpret_cast<const uint4*>(p);
}

__device__ __forceinline__ float rowMax16(float v) {
    v = fmaxf(v, __shfl_xor(v, 1, 16));
    v = fmaxf(v, __shfl_xor(v, 2, 16));
    v = fmaxf(v, __shfl_xor(v, 4, 16));
    v = fmaxf(v, __shfl_xor(v, 8, 16));
    return v;
}
__device__ __forceinline__ float rowSum16(float v) {
    v += __shfl_xor(v, 1, 16);
    v += __shfl_xor(v, 2, 16);
    v += __shfl_xor(v, 4, 16);
    v += __shfl_xor(v, 8, 16);
    return v;
}

// ---------------------------------------------------------------------------
// fp32 -> bf16 bulk convert (8 elements / thread, b128 in, b128 out)
// ---------------------------------------------------------------------------
__global__ __launch_bounds__(256)
void convert_x(const float* __restrict__ x, bf16_t* __restrict__ y)
{
    const size_t i = (size_t)blockIdx.x * 256 + threadIdx.x;   // 8 elems each
    const float4* xv = reinterpret_cast<const float4*>(x);
    const float4 a = xv[2 * i], b = xv[2 * i + 1];
    union { uint4 u; bf16_t h[8]; } pk;
    pk.h[0] = (bf16_t)a.x; pk.h[1] = (bf16_t)a.y;
    pk.h[2] = (bf16_t)a.z; pk.h[3] = (bf16_t)a.w;
    pk.h[4] = (bf16_t)b.x; pk.h[5] = (bf16_t)b.y;
    pk.h[6] = (bf16_t)b.z; pk.h[7] = (bf16_t)b.w;
    reinterpret_cast<uint4*>(y)[i] = pk.u;
}

// ---------------------------------------------------------------------------
// W[k][n] fp32 -> Wt[n][k] bf16, LDS-tiled 32x32 transpose (coalesced both ways)
// grid (DMODEL/32, DMODEL/32), 256 thr
// ---------------------------------------------------------------------------
__global__ __launch_bounds__(256)
void convert_wt(const float* __restrict__ w, bf16_t* __restrict__ wt)
{
    __shared__ float tile[32][33];
    const int kBase = blockIdx.x * 32;
    const int nBase = blockIdx.y * 32;
    const int tx = threadIdx.x & 31;
    const int ty = threadIdx.x >> 5;      // 0..7
#pragma unroll
    for (int i = 0; i < 4; ++i)
        tile[ty + 8 * i][tx] = w[(size_t)(kBase + ty + 8 * i) * DMODEL + nBase + tx];
    __syncthreads();
#pragma unroll
    for (int i = 0; i < 4; ++i)
        wt[(size_t)(nBase + ty + 8 * i) * DMODEL + kBase + tx] =
            (bf16_t)tile[tx][ty + 8 * i];
}

// ---------------------------------------------------------------------------
// LDS-free GEMM: Y[4096x768] = X[4096x768](bf16) * Wt[n][k](bf16)^T + bias
// Block 256 thr = 8 waves (4M x 2N); wave tile 32x64 -> block tile 128x128.
// Operands stream from L2; 8 WMMA per wave per 32-wide k-step, no barriers.
// MODE 0/1: bf16 [b,h,s,d]   MODE 2: bf16 [b,h,d,s]   MODE 3: f32 [tok,D]
// ---------------------------------------------------------------------------
template <int MODE>
__global__ __launch_bounds__(256)
void gemm_bf16_direct(const bf16_t* __restrict__ xa, const bf16_t* __restrict__ wt,
                      const float* __restrict__ bias, void* __restrict__ out)
{
    const int lane  = threadIdx.x & 31;
    const int wid   = threadIdx.x >> 5;
    const int waveM = wid >> 1;                       // 0..3
    const int waveN = wid & 1;                        // 0..1
    const int rowBase = blockIdx.x * 128 + waveM * 32;
    const int colBase = blockIdx.y * 128 + waveN * 64;
    const int l15 = lane & 15;
    const int k0  = (lane < 16) ? 0 : 8;              // 16-bit A-layout K split

    const v8f z8 = {0.f,0.f,0.f,0.f,0.f,0.f,0.f,0.f};
    v8f acc[2][4] = {{z8, z8, z8, z8}, {z8, z8, z8, z8}};

    const bf16_t* arow0 = xa + (size_t)(rowBase + l15) * DMODEL;
    const bf16_t* arow1 = arow0 + (size_t)16 * DMODEL;
    const bf16_t* brow[4];
#pragma unroll
    for (int nt = 0; nt < 4; ++nt)
        brow[nt] = wt + (size_t)(colBase + nt * 16 + l15) * DMODEL;

    for (int ks = 0; ks < DMODEL; ks += 32) {
        FragBF fa[2], fb[4];
        fa[0].u[0] = ld16g(arow0 + ks + k0);
        fa[0].u[1] = ld16g(arow0 + ks + k0 + 16);
        fa[1].u[0] = ld16g(arow1 + ks + k0);
        fa[1].u[1] = ld16g(arow1 + ks + k0 + 16);
#pragma unroll
        for (int nt = 0; nt < 4; ++nt) {
            fb[nt].u[0] = ld16g(brow[nt] + ks + k0);
            fb[nt].u[1] = ld16g(brow[nt] + ks + k0 + 16);
        }
#pragma unroll
        for (int mt = 0; mt < 2; ++mt)
#pragma unroll
            for (int nt = 0; nt < 4; ++nt)
                acc[mt][nt] = wmma_bf16(fa[mt].v, fb[nt].v, acc[mt][nt]);
    }

    // epilogue: bias + layout-specific store
#pragma unroll
    for (int mt = 0; mt < 2; ++mt) {
#pragma unroll
        for (int nt = 0; nt < 4; ++nt) {
            const int gn   = colBase + nt * 16 + l15;
            const float bc = bias[gn];
            const int gmB  = rowBase + mt * 16 + ((lane >> 4) << 3);
#pragma unroll
            for (int r = 0; r < 8; ++r) {
                const int gm = gmB + r;
                const float val = acc[mt][nt][r] + bc;
                if constexpr (MODE == 0 || MODE == 1) {
                    const int b = gm >> 11, s = gm & 2047;
                    const int h = gn >> 6,  d = gn & 63;
                    ((bf16_t*)out)[(((size_t)(b * NH + h)) * SEQ + s) * HD + d] =
                        (bf16_t)val;
                } else if constexpr (MODE == 2) {
                    const int b = gm >> 11, s = gm & 2047;
                    const int h = gn >> 6,  d = gn & 63;
                    ((bf16_t*)out)[(((size_t)(b * NH + h)) * HD + d) * SEQ + s] =
                        (bf16_t)val;
                } else {
                    ((float*)out)[(size_t)gm * DMODEL + gn] = val;
                }
            }
        }
    }
}

// ---------------------------------------------------------------------------
// Fused flash attention. Block = 128 thr (4 waves). Each wave: 16 queries of
// one (b,h), streams 32-key tiles over S=2048 directly from L2-resident K/V.
// grid = (S/64, B*H)
// ---------------------------------------------------------------------------
__global__ __launch_bounds__(128)
void attention_fused(const bf16_t* __restrict__ qg, const bf16_t* __restrict__ kg,
                     const bf16_t* __restrict__ vtg, bf16_t* __restrict__ og)
{
    __shared__ bf16_t pl[4][16][40];   // per-wave P relayout scratch (bf16)

    const int lane  = threadIdx.x & 31;
    const int wid   = threadIdx.x >> 5;
    const int bh    = blockIdx.y;                       // b*NH + h
    const int qtile = blockIdx.x * 64 + wid * 16;       // s of first query
    const int l15   = lane & 15;
    const int k0    = (lane < 16) ? 0 : 8;

    const bf16_t* q  = qg  + (size_t)bh * SEQ * HD;
    const bf16_t* k  = kg  + (size_t)bh * SEQ * HD;
    const bf16_t* vt = vtg + (size_t)bh * HD * SEQ;

    FragBF aq[2];
    const int qrow = qtile + l15;
#pragma unroll
    for (int t = 0; t < 2; ++t) {
        aq[t].u[0] = ld16g(q + (size_t)qrow * HD + t * 32 + k0);
        aq[t].u[1] = ld16g(q + (size_t)qrow * HD + t * 32 + k0 + 16);
    }

    const v8f z8 = {0.f,0.f,0.f,0.f,0.f,0.f,0.f,0.f};
    v8f o[4] = {z8, z8, z8, z8};
    float mrow[8], lrow[8];
#pragma unroll
    for (int r = 0; r < 8; ++r) { mrow[r] = -1e30f; lrow[r] = 0.f; }

    const float scale = 0.125f;                  // 1/sqrt(64)

    for (int kt = 0; kt < SEQ; kt += 32) {
        if (kt + 32 < SEQ) {                     // L2 prefetch of next tiles
            __builtin_prefetch(k + (size_t)(kt + 32 + l15) * HD, 0, 1);
            __builtin_prefetch(vt + (size_t)l15 * SEQ + kt + 32, 0, 1);
        }
        // scores: two 16x16 C tiles over 32 keys, K-dim = head dim 64
        v8f c0 = z8, c1 = z8;
        {
            FragBF bk;
            const int kr0 = kt + l15;
            bk.u[0] = ld16g(k + (size_t)kr0 * HD + k0);
            bk.u[1] = ld16g(k + (size_t)kr0 * HD + k0 + 16);
            c0 = wmma_bf16(aq[0].v, bk.v, c0);
            bk.u[0] = ld16g(k + (size_t)kr0 * HD + 32 + k0);
            bk.u[1] = ld16g(k + (size_t)kr0 * HD + 32 + k0 + 16);
            c0 = wmma_bf16(aq[1].v, bk.v, c0);

            const int kr1 = kt + 16 + l15;
            bk.u[0] = ld16g(k + (size_t)kr1 * HD + k0);
            bk.u[1] = ld16g(k + (size_t)kr1 * HD + k0 + 16);
            c1 = wmma_bf16(aq[0].v, bk.v, c1);
            bk.u[0] = ld16g(k + (size_t)kr1 * HD + 32 + k0);
            bk.u[1] = ld16g(k + (size_t)kr1 * HD + 32 + k0 + 16);
            c1 = wmma_bf16(aq[1].v, bk.v, c1);
        }

        // online softmax; width-16 reductions act per C-layout row
        float alpha[8];
#pragma unroll
        for (int r = 0; r < 8; ++r) {
            const float s0 = c0[r] * scale;
            const float s1 = c1[r] * scale;
            const float rm = rowMax16(fmaxf(s0, s1));
            const float mn = fmaxf(mrow[r], rm);
            alpha[r]       = __expf(mrow[r] - mn);
            const float p0 = __expf(s0 - mn);
            const float p1 = __expf(s1 - mn);
            lrow[r] = lrow[r] * alpha[r] + rowSum16(p0 + p1);
            mrow[r] = mn;
            const int prow = r + ((lane >> 4) << 3);
            pl[wid][prow][l15]      = (bf16_t)p0;
            pl[wid][prow][16 + l15] = (bf16_t)p1;
        }
#pragma unroll
        for (int t = 0; t < 4; ++t)
#pragma unroll
            for (int r = 0; r < 8; ++r) o[t][r] *= alpha[r];

        // P (C-layout f32) -> A-layout bf16 via wave-local LDS
        FragBF ap;
        ap.u[0] = *(const uint4*)&pl[wid][l15][k0];
        ap.u[1] = *(const uint4*)&pl[wid][l15][k0 + 16];

        // O += P * V   (V^T rows contiguous over keys)
#pragma unroll
        for (int t = 0; t < 4; ++t) {
            FragBF bv;
            const int d = t * 16 + l15;
            bv.u[0] = ld16g(vt + (size_t)d * SEQ + kt + k0);
            bv.u[1] = ld16g(vt + (size_t)d * SEQ + kt + k0 + 16);
            o[t] = wmma_bf16(ap.v, bv.v, o[t]);
        }
    }

    // normalize + store merged-head bf16 [tok, D]
    const int hIdx = bh % NH;
    const int bIdx = bh / NH;
#pragma unroll
    for (int t = 0; t < 4; ++t) {
#pragma unroll
        for (int r = 0; r < 8; ++r) {
            const int m   = r + ((lane >> 4) << 3);
            const int s   = qtile + m;
            const int col = hIdx * HD + t * 16 + l15;
            const float val = o[t][r] / lrow[r];
            og[((size_t)(bIdx * SEQ + s)) * DMODEL + col] = (bf16_t)val;
        }
    }
}

// ---------------------------------------------------------------------------
extern "C" void kernel_launch(void* const* d_in, const int* in_sizes, int n_in,
                              void* d_out, int out_size, void* d_ws, size_t ws_size,
                              hipStream_t stream)
{
    (void)in_sizes; (void)n_in; (void)out_size; (void)ws_size;
    const float* query = (const float*)d_in[0];
    const float* key_  = (const float*)d_in[1];
    const float* value = (const float*)d_in[2];
    const float* wq = (const float*)d_in[3];
    const float* bq = (const float*)d_in[4];
    const float* wk = (const float*)d_in[5];
    const float* bk = (const float*)d_in[6];
    const float* wv = (const float*)d_in[7];
    const float* bv = (const float*)d_in[8];
    const float* wo = (const float*)d_in[9];
    const float* bo = (const float*)d_in[10];

    const size_t P   = (size_t)NTOK * DMODEL * sizeof(bf16_t);   // 6.29 MB
    const size_t WSZ = (size_t)DMODEL * DMODEL * sizeof(bf16_t); // 1.18 MB
    char* base = (char*)d_ws;
    bf16_t* qb   = (bf16_t*)(base + 0 * P);
    bf16_t* kb   = (bf16_t*)(base + 1 * P);
    bf16_t* vt   = (bf16_t*)(base + 2 * P);
    bf16_t* xq   = (bf16_t*)(base + 3 * P);   // reused as attn after proj
    bf16_t* xk   = (bf16_t*)(base + 4 * P);
    bf16_t* xv   = (bf16_t*)(base + 5 * P);
    bf16_t* attn = xq;
    bf16_t* wqt  = (bf16_t*)(base + 6 * P + 0 * WSZ);
    bf16_t* wkt  = (bf16_t*)(base + 6 * P + 1 * WSZ);
    bf16_t* wvt  = (bf16_t*)(base + 6 * P + 2 * WSZ);
    bf16_t* wot  = (bf16_t*)(base + 6 * P + 3 * WSZ);

    // 1) one-time conversions (activations + transposed weights)
    const int cxBlocks = (NTOK * DMODEL) / (256 * 8);              // 1536
    convert_x<<<cxBlocks, 256, 0, stream>>>(query, xq);
    convert_x<<<cxBlocks, 256, 0, stream>>>(key_,  xk);
    convert_x<<<cxBlocks, 256, 0, stream>>>(value, xv);
    const dim3 wtGrid(DMODEL / 32, DMODEL / 32);                   // 24x24
    convert_wt<<<wtGrid, 256, 0, stream>>>(wq, wqt);
    convert_wt<<<wtGrid, 256, 0, stream>>>(wk, wkt);
    convert_wt<<<wtGrid, 256, 0, stream>>>(wv, wvt);
    convert_wt<<<wtGrid, 256, 0, stream>>>(wo, wot);

    // 2) LDS-free WMMA projections (operands L2-resident)
    const dim3 gGrid(NTOK / 128, DMODEL / 128);                    // 32 x 6
    gemm_bf16_direct<0><<<gGrid, 256, 0, stream>>>(xq, wqt, bq, qb);
    gemm_bf16_direct<1><<<gGrid, 256, 0, stream>>>(xk, wkt, bk, kb);
    gemm_bf16_direct<2><<<gGrid, 256, 0, stream>>>(xv, wvt, bv, vt);

    // 3) fused flash attention (xq is dead -> attn aliases it)
    attention_fused<<<dim3(SEQ / 64, NB * NH), 128, 0, stream>>>(qb, kb, vt, attn);

    // 4) output projection, fp32 + bias to d_out
    gemm_bf16_direct<3><<<gGrid, 256, 0, stream>>>(attn, wot, bo, d_out);
}
